// Net_90022514524410
// MI455X (gfx1250) — compile-verified, matrix operand order checked by reading, not run on hardware
//
#include <hip/hip_runtime.h>
#include <hip/hip_bf16.h>

typedef __attribute__((ext_vector_type(16))) __bf16 v16bf;
typedef __attribute__((ext_vector_type(8)))  float  v8f;

#define T_SEQ   256
#define E_DIM   50
#define KPAD    64          // K padded to 2 x 32 (two WMMA K-slices)
#define NGATE   200         // 4*E
#define NPAD    208         // 13 tiles of 16
#define NT_TILES 13
#define MT      16          // batch rows per workgroup (one WMMA M tile)
#define WAVES   8
#define TPW     2           // N-tiles per wave (8*2 = 16 >= 13)

// ISA 7.12.2: 16-bit A/B fragment element e (0..15) -> K index within a K=32 slice.
__device__ __forceinline__ int kmap(int e, int hi) {
  return (e & 7) + ((e >> 3) << 4) + (hi << 3);
}

// Load an A fragment (16x32 bf16) from an LDS tile stored row-major [16][KPAD].
// Pairwise-contiguous K mapping lets the compiler coalesce to ds_load_b128.
__device__ __forceinline__ v16bf load_a(const __bf16* base, int ks, int m, int hi) {
  v16bf a;
#pragma unroll
  for (int e = 0; e < 16; ++e) a[e] = base[m * KPAD + ks + kmap(e, hi)];
  return a;
}

// Load a B fragment (32x16) straight from the f32 weight matrix W[NGATE][E_DIM],
// B[k][n] = W[n][k], zero-padding K>=50 and N>=200. Time-invariant -> registers.
__device__ __forceinline__ v16bf load_b(const float* __restrict__ W, int n0,
                                        int ks, int nl, int hi) {
  v16bf b;
  int n = n0 + nl;
#pragma unroll
  for (int e = 0; e < 16; ++e) {
    int k = ks + kmap(e, hi);
    float v = (n < NGATE && k < E_DIM) ? W[n * E_DIM + k] : 0.f;
    b[e] = (__bf16)v;
  }
  return b;
}

// Fast activations: v_mul + v_exp_f32 + v_add + v_rcp_f32 (no IEEE divide path).
__device__ __forceinline__ float sigmoid_fast(float x) {
  return __builtin_amdgcn_rcpf(1.f + __expf(-x));
}
__device__ __forceinline__ float tanh_fast(float x) {
  return 2.f * sigmoid_fast(2.f * x) - 1.f;   // (1 - e^-2x) / (1 + e^-2x)
}

__global__ __launch_bounds__(WAVES * 32)
void lstm_fused_kernel(const int* __restrict__ sent, const int* __restrict__ mask,
                       const float* __restrict__ emb, const float* __restrict__ Wih,
                       const float* __restrict__ Whh, const float* __restrict__ bih,
                       const float* __restrict__ bhh, float* __restrict__ out, int B) {
  __shared__ __bf16 sx[MT * KPAD];   // x_t tile, bf16, zero-padded K
  __shared__ __bf16 sh[MT * KPAD];   // h_t tile, bf16, zero-padded K
  __shared__ float  shf[MT * KPAD];  // h_t tile, f32 (exact state)
  __shared__ float  sc[MT * KPAD];   // c_t tile, f32
  __shared__ float  sg[MT * NPAD];   // gate tile (f32 accum spill)
  __shared__ float  sbias[NPAD];     // b_ih + b_hh
  __shared__ int    stok[MT];
  __shared__ int    slen[MT];

  const int tid  = threadIdx.x;
  const int lane = tid & 31;
  const int wave = tid >> 5;
  const int m16  = lane & 15;  // M (for A) or N-local (for B/acc/D)
  const int hi   = lane >> 4;
  const int b0   = blockIdx.x * MT;

  // ---- one-time init: zero state, fold biases, sequence lengths ----
  for (int i = tid; i < MT * KPAD; i += WAVES * 32) {
    sx[i] = (__bf16)0.f; sh[i] = (__bf16)0.f; shf[i] = 0.f; sc[i] = 0.f;
  }
  for (int i = tid; i < NPAD; i += WAVES * 32)
    sbias[i] = (i < NGATE) ? (bih[i] + bhh[i]) : 0.f;
  if (tid < MT) {
    int row = b0 + tid, L = 0;
    if (row < B) {
      const int* mr = mask + (size_t)row * T_SEQ;
      for (int tt = 0; tt < T_SEQ; ++tt) L += mr[tt];
    }
    slen[tid] = L;
  }

  // ---- preload time-invariant B fragments into registers ----
  v16bf Bw[2][TPW][2];  // [Wih|Whh][tile slot][K slice]
#pragma unroll
  for (int s = 0; s < TPW; ++s) {
    int n0 = (wave * TPW + s) * 16;
#pragma unroll
    for (int ks = 0; ks < 2; ++ks) {
      Bw[0][s][ks] = load_b(Wih, n0, ks * 32, m16, hi);
      Bw[1][s][ks] = load_b(Whh, n0, ks * 32, m16, hi);
    }
  }
  __syncthreads();

  // ---- recurrence ----
  for (int t = 0; t < T_SEQ; ++t) {
    // tokens for this step, prefetch next step's embedding rows toward L0/L2
    if (tid < MT) {
      int row = b0 + tid;
      int tok = (row < B) ? sent[(size_t)row * T_SEQ + t] : 0;
      stok[tid] = tok;
      if (row < B && (t + 1) < T_SEQ) {
        int tn = sent[(size_t)row * T_SEQ + t + 1];
        __builtin_prefetch(&emb[(size_t)tn * E_DIM], 0, 0);
      }
    }
    __syncthreads();

    // gather x_t (embedding rows stay L2-resident: table is 10 MB vs 192 MB L2)
    for (int i = tid; i < MT * KPAD; i += WAVES * 32) {
      int m = i >> 6, k = i & (KPAD - 1);
      float v = (k < E_DIM) ? emb[(size_t)stok[m] * E_DIM + k] : 0.f;
      sx[i] = (__bf16)v;
    }
    __syncthreads();

    // A fragments are identical for every N-tile this wave owns: load once.
    v16bf ax0 = load_a(sx, 0,  m16, hi);
    v16bf ax1 = load_a(sx, 32, m16, hi);
    v16bf ah0 = load_a(sh, 0,  m16, hi);
    v16bf ah1 = load_a(sh, 32, m16, hi);

    // gates = bias + x_t @ Wih^T + h @ Whh^T   (4 chained WMMAs per N-tile)
#pragma unroll
    for (int s = 0; s < TPW; ++s) {
      int nt = wave * TPW + s;          // wave-uniform: EXEC stays all-ones
      if (nt < NT_TILES) {
        int n0 = nt * 16;
        float bv = sbias[n0 + m16];
        v8f acc;
#pragma unroll
        for (int r = 0; r < 8; ++r) acc[r] = bv;
        acc = __builtin_amdgcn_wmma_f32_16x16x32_bf16(false, ax0, false, Bw[0][s][0], (short)0, acc, false, false);
        acc = __builtin_amdgcn_wmma_f32_16x16x32_bf16(false, ax1, false, Bw[0][s][1], (short)0, acc, false, false);
        acc = __builtin_amdgcn_wmma_f32_16x16x32_bf16(false, ah0, false, Bw[1][s][0], (short)0, acc, false, false);
        acc = __builtin_amdgcn_wmma_f32_16x16x32_bf16(false, ah1, false, Bw[1][s][1], (short)0, acc, false, false);
        // D layout: lanes 0-15 N=lane, M=r ; lanes 16-31 N=lane-16, M=r+8
#pragma unroll
        for (int r = 0; r < 8; ++r)
          sg[(r + hi * 8) * NPAD + n0 + m16] = acc[r];
      }
    }
    __syncthreads();

    // masked f32 cell update; freeze state once t >= length
    for (int i = tid; i < MT * E_DIM; i += WAVES * 32) {
      int m = i / E_DIM, e = i - m * E_DIM;
      if (t < slen[m]) {
        const float* gm = sg + m * NPAD;
        float ig = sigmoid_fast(gm[e]);
        float fg = sigmoid_fast(gm[E_DIM + e]);
        float cg = tanh_fast(gm[2 * E_DIM + e]);
        float og = sigmoid_fast(gm[3 * E_DIM + e]);
        float cn = fg * sc[m * KPAD + e] + ig * cg;
        float hn = og * tanh_fast(cn);
        sc [m * KPAD + e] = cn;
        shf[m * KPAD + e] = hn;
        sh [m * KPAD + e] = (__bf16)hn;   // bf16 copy feeds next step's WMMA A
      }
    }
    __syncthreads();
  }

  // ---- emit h_last [B, E] f32 ----
  for (int i = tid; i < MT * E_DIM; i += WAVES * 32) {
    int m = i / E_DIM, e = i - m * E_DIM;
    int row = b0 + m;
    if (row < B) out[(size_t)row * E_DIM + e] = shf[m * KPAD + e];
  }
}

extern "C" void kernel_launch(void* const* d_in, const int* in_sizes, int n_in,
                              void* d_out, int out_size, void* d_ws, size_t ws_size,
                              hipStream_t stream) {
  const int*   sent = (const int*)d_in[0];
  const int*   mask = (const int*)d_in[1];
  const float* emb  = (const float*)d_in[2];
  const float* Wih  = (const float*)d_in[3];
  const float* Whh  = (const float*)d_in[4];
  const float* bih  = (const float*)d_in[5];
  const float* bhh  = (const float*)d_in[6];
  float* out = (float*)d_out;
  int B = in_sizes[0] / T_SEQ;          // sent is [B, T]
  int grid = (B + MT - 1) / MT;         // 512 independent workgroups at B=8192
  lstm_fused_kernel<<<grid, WAVES * 32, 0, stream>>>(sent, mask, emb, Wih, Whh,
                                                     bih, bhh, out, B);
}
